// GCBFGNN_27358941675625
// MI455X (gfx1250) — matrix operand order repaired and checked by reading, not compiled, not run on hardware
//
#include <hip/hip_runtime.h>
#include <cstddef>
#include <cstdint>

// ---------------------------------------------------------------------------
// GAT-GNN (N=1024, H=4, HIDDEN=[128,128,64]) for MI455X (gfx1250, wave32).
//
// edges@We is folded into per-node vectors t[i,h,c] = sum_d q[i,h,d]*We[c,hd*h+d]
// so the edge-conditioned score is an 8-dim dot per (i,j,h).
//
// Fused flash-style attention: one wave owns (i-tile of 16, all 4 heads,
// j-chunk of 64). S^T = K*Q^T via v_wmma_f32_16x16x32_f16 so the D-layout
// (lane = fixed i, 8 j's per VGPR) maps directly onto the A-layout of P for
// the P*V WMMA with no cross-lane movement. No-max softmax => j-chunk
// partials combine by plain addition (deterministic, no atomics).
//
// Scheduling (by construction, compile-only): all K/V tiles for a j-iteration
// are batch-loaded up front from running base pointers with immediate offsets,
// so VMEM latency overlaps the VALU-heavy edge-bias/exp section and the WMMA
// block instead of serializing on s_wait_loadcnt 0 per tile.
// ---------------------------------------------------------------------------

typedef __attribute__((ext_vector_type(16))) _Float16 v16h;
typedef __attribute__((ext_vector_type(8)))  _Float16 v8h;
typedef __attribute__((ext_vector_type(8)))  float    v8f;

#define NN    1024
#define NHEAD 4
#define NJC   16                // j-chunks (partial-softmax splits)
#define JCHUNK (NN / NJC)       // 64 j's per wave -> 2 iterations of 32

// ---------------------------------------------------------------------------
// Generic dense: y[i,o] = relu?(sum_k x[i,k]*W[k,o] + b[o])
// ---------------------------------------------------------------------------
__global__ void dense_kernel(const float* __restrict__ x,
                             const float* __restrict__ W,
                             const float* __restrict__ b,
                             int IN, int OUT, int do_relu,
                             float* __restrict__ y) {
    int t = blockIdx.x * blockDim.x + threadIdx.x;
    if (t >= NN * OUT) return;
    int i = t / OUT, o = t % OUT;
    float s = b[o];
    for (int k = 0; k < IN; ++k) s += x[(size_t)i * IN + k] * W[(size_t)k * OUT + o];
    if (do_relu) s = fmaxf(s, 0.0f);
    y[t] = s;
}

// ---------------------------------------------------------------------------
// QKV projection; q kept f32 for the We-folding, q/k as f16 row-major
// [N, H*hd], v transposed f16 [H*hd, N] (B-layout friendly for P*V WMMA).
// ---------------------------------------------------------------------------
__global__ void qkv_kernel(const float* __restrict__ x, int IN,
                           const float* __restrict__ Wq, const float* __restrict__ bq,
                           const float* __restrict__ Wk, const float* __restrict__ bk,
                           const float* __restrict__ Wv, const float* __restrict__ bv,
                           int HDIM,
                           float* __restrict__ qf,
                           _Float16* __restrict__ q16,
                           _Float16* __restrict__ k16,
                           _Float16* __restrict__ v16T) {
    int t = blockIdx.x * blockDim.x + threadIdx.x;
    if (t >= NN * HDIM) return;
    int i = t / HDIM, o = t % HDIM;
    float q = bq[o], k = bk[o], v = bv[o];
    for (int c = 0; c < IN; ++c) {
        float xv = x[(size_t)i * IN + c];
        q += xv * Wq[(size_t)c * HDIM + o];
        k += xv * Wk[(size_t)c * HDIM + o];
        v += xv * Wv[(size_t)c * HDIM + o];
    }
    qf[t]  = q;
    q16[t] = (_Float16)q;
    k16[t] = (_Float16)k;
    v16T[(size_t)o * NN + i] = (_Float16)v;
}

// ---------------------------------------------------------------------------
// Fold We (and its bias) into per-node vectors:
//   t[i,h,c] = sum_d q[i,h,d] * We[c, h*hd+d];  sb[i,h] = sum_d q[i,h,d]*be[..]
// ---------------------------------------------------------------------------
__global__ void tq_kernel(const float* __restrict__ qf,
                          const float* __restrict__ We,
                          const float* __restrict__ be,
                          int HD,
                          float* __restrict__ tq, float* __restrict__ sbq) {
    int t = blockIdx.x * blockDim.x + threadIdx.x;
    if (t >= NN * NHEAD) return;
    int i = t / NHEAD, h = t % NHEAD;
    int HDIM = NHEAD * HD;
    float acc[8] = {0,0,0,0,0,0,0,0};
    float sb = 0.0f;
    for (int d = 0; d < HD; ++d) {
        float qv = qf[(size_t)i * HDIM + h * HD + d];
#pragma unroll
        for (int c = 0; c < 8; ++c) acc[c] += qv * We[(size_t)c * HDIM + h * HD + d];
        sb += qv * be[h * HD + d];
    }
#pragma unroll
    for (int c = 0; c < 8; ++c) tq[((size_t)i * NHEAD + h) * 8 + c] = acc[c];
    sbq[t] = sb;
}

// ---------------------------------------------------------------------------
// Fused attention (flash-style, no-max softmax, partial over j-chunks).
//
// Assumed gfx1250 WMMA wave32 layouts (16x16x32 f16):
//   A (16Mx32K): lane l: M=l&15; halfs 0..7 -> K = 8*(l>>4)+0..7,
//                               halfs 8..15 -> K = 16+8*(l>>4)+0..7
//   B (32Kx16N): lane l: N=l&15; same K order as A (vB uses it); kA and qB both use K = 16*(l>>4)+0..15
//   C/D (16x16 f32): lane l: N=l&15; VGPR v -> M = v + 8*(l>>4)
// ---------------------------------------------------------------------------
template<int HD>
__global__ __launch_bounds__(256)
void gat_attn_kernel(const float* __restrict__ edges,   // [N,N,8]
                     const int* __restrict__ adj,       // [N,N]
                     const _Float16* __restrict__ q16,  // [N, 4*HD]
                     const _Float16* __restrict__ k16,  // [N, 4*HD]
                     const _Float16* __restrict__ v16T, // [4*HD, N]
                     const float* __restrict__ tq,      // [N,4,8]
                     const float* __restrict__ sbq,     // [N,4]
                     float scale,
                     float* __restrict__ Opart,         // [NJC,N,4,HD]
                     float* __restrict__ Lpart)         // [NJC,N,4]
{
    const int lane = threadIdx.x & 31;
    const int wave = blockIdx.x * (blockDim.x >> 5) + (threadIdx.x >> 5);
    const int it = wave / NJC;            // i-tile index (0..63)
    const int jc = wave % NJC;            // j-chunk index
    const int ibase = it * 16;
    const int l15 = lane & 15;
    const int hi  = lane >> 4;
    const int i   = ibase + l15;          // this lane's attention row
    const int HDIM = NHEAD * HD;

    // ---- Q in B-layout (K-dim = d), per head ----
    v16h qB[NHEAD];
    {
        v16h zz = {};
#pragma unroll
        for (int h = 0; h < NHEAD; ++h) {
            if (HD == 32) {
                qB[h] = *(const v16h*)(q16 + ((size_t)i * NHEAD + h) * HD + hi * 16);
            } else {
                qB[h] = zz;
                if (hi == 0) qB[h] = *(const v16h*)(q16 + ((size_t)i * NHEAD + h) * HD);
            }
        }
    }

    // ---- per-lane folded edge weights & bias, pre-scaled by 1/sqrt(hd) ----
    float tt[NHEAD][8];
    float sb[NHEAD];
#pragma unroll
    for (int h = 0; h < NHEAD; ++h) {
#pragma unroll
        for (int c = 0; c < 8; ++c) tt[h][c] = tq[((size_t)i * NHEAD + h) * 8 + c] * scale;
        sb[h] = sbq[(size_t)i * NHEAD + h] * scale;
    }

    v8f acc[NHEAD][HD / 16];
    {
        v8f z = {};
#pragma unroll
        for (int h = 0; h < NHEAD; ++h)
#pragma unroll
            for (int dc = 0; dc < HD / 16; ++dc) acc[h][dc] = z;
    }
    float lsum[NHEAD] = {0.f, 0.f, 0.f, 0.f};

    const int j0beg = jc * JCHUNK;
    // Running base pointers (advanced once per 32-j step; tile accesses below
    // are compile-time immediate offsets from these).
    const _Float16* kptr = k16 + (size_t)(j0beg + l15) * HDIM + 16 * hi;
    const _Float16* vptr = v16T + (size_t)l15 * NN + j0beg + 8 * hi;
    const float*    eptr = edges + ((size_t)i * NN + j0beg + 8 * hi) * 8;
    const int*      aptr = adj + (size_t)i * NN + j0beg + 8 * hi;

    for (int iter = 0; iter < JCHUNK / 32; ++iter) {
        // ---- 1) batch-load all K tiles (A-layout) ----
        v16h kA[NHEAD][2];
#pragma unroll
        for (int h = 0; h < NHEAD; ++h) {
#pragma unroll
            for (int s = 0; s < 2; ++s) {
                const _Float16* kp = kptr + (size_t)h * HD + (size_t)s * 16 * HDIM;
                v8h lo, hh;
                if (HD == 32) {
                    lo = *(const v8h*)(kp);
                    hh = *(const v8h*)(kp + 8);
                } else {
                    v8h zh = {};
                    lo = *(const v8h*)(kp);
                    hh = *(const v8h*)(kp + 8); (void)zh;
                }
#pragma unroll
                for (int e = 0; e < 8; ++e) { kA[h][s][e] = lo[e]; kA[h][s][e + 8] = hh[e]; }
            }
        }

        // ---- 2) batch-load all V tiles (B-layout); in flight across the
        //         entire VALU section below ----
        v16h vB[NHEAD][HD / 16];
#pragma unroll
        for (int h = 0; h < NHEAD; ++h)
#pragma unroll
            for (int dc = 0; dc < HD / 16; ++dc)
            {
                const _Float16* vp = vptr + (size_t)(h * HD + dc * 16) * NN;
                v8h vlo = *(const v8h*)(vp);
                v8h vhi = *(const v8h*)(vp + 16);
#pragma unroll
                for (int e = 0; e < 8; ++e) { vB[h][dc][e] = vlo[e]; vB[h][dc][e + 8] = vhi[e]; }
            }

        // ---- 3) S^T = K * Q^T ----
        v8f st[NHEAD][2];
#pragma unroll
        for (int h = 0; h < NHEAD; ++h) {
#pragma unroll
            for (int s = 0; s < 2; ++s) {
                v8f cz = {};
                st[h][s] = __builtin_amdgcn_wmma_f32_16x16x32_f16(
                    false, kA[h][s], false, qB[h], (short)0, cz, false, false);
            }
        }

        // ---- 4) edge bias + mask + exp + pack P (A-layout, no lane moves) ----
        __builtin_prefetch(eptr + 32 * 8, 0, 1);        // next 32-j tile
        __builtin_prefetch(eptr + 32 * 8 + 128, 0, 1);
        v16h pA[NHEAD];
#pragma unroll
        for (int s = 0; s < 2; ++s) {
#pragma unroll
            for (int v = 0; v < 8; ++v) {
                const float* ep = eptr + (s * 16 + v) * 8;
                float4 ea = *(const float4*)ep;
                float4 eb = *(const float4*)(ep + 4);
                int a = aptr[s * 16 + v];
#pragma unroll
                for (int h = 0; h < NHEAD; ++h) {
                    // pre-scaled: ed = scale*(sum_c t*e + sb_raw)
                    float ed = sb[h];
                    ed = fmaf(tt[h][0], ea.x, ed); ed = fmaf(tt[h][1], ea.y, ed);
                    ed = fmaf(tt[h][2], ea.z, ed); ed = fmaf(tt[h][3], ea.w, ed);
                    ed = fmaf(tt[h][4], eb.x, ed); ed = fmaf(tt[h][5], eb.y, ed);
                    ed = fmaf(tt[h][6], eb.z, ed); ed = fmaf(tt[h][7], eb.w, ed);
                    float sc = fmaf(st[h][s][v], scale, ed);
                    float p  = a ? __expf(sc) : 0.0f;   // masked -> exactly 0
                    lsum[h] += p;
                    pA[h][s * 8 + v] = (_Float16)p;
                }
            }
        }

        // ---- 5) O += P * V ----
#pragma unroll
        for (int h = 0; h < NHEAD; ++h)
#pragma unroll
            for (int dc = 0; dc < HD / 16; ++dc)
                acc[h][dc] = __builtin_amdgcn_wmma_f32_16x16x32_f16(
                    false, pA[h], false, vB[h][dc], (short)0, acc[h][dc], false, false);

        kptr += 32 * HDIM;
        vptr += 32;
        eptr += 32 * 8;
        aptr += 32;
    }

    // ---- write partials (disjoint per (jc,i,h,d): deterministic) ----
#pragma unroll
    for (int h = 0; h < NHEAD; ++h) {
        float l2 = lsum[h] + __shfl_xor(lsum[h], 16, 32);  // lanes l & l+16 share i
        if (hi == 0) Lpart[((size_t)jc * NN + i) * NHEAD + h] = l2;
#pragma unroll
        for (int dc = 0; dc < HD / 16; ++dc) {
#pragma unroll
            for (int v = 0; v < 8; ++v) {
                int irow = ibase + v + 8 * hi;
                Opart[(((size_t)jc * NN + irow) * NHEAD + h) * HD + dc * 16 + l15] =
                    acc[h][dc][v];
            }
        }
    }
}

// ---------------------------------------------------------------------------
// Combine j-chunk partials: agg[i, h*HD+d] = sum_jc O / max(sum_jc L, eps)
// ---------------------------------------------------------------------------
__global__ void gat_reduce_kernel(const float* __restrict__ Opart,
                                  const float* __restrict__ Lpart,
                                  int HD, float* __restrict__ agg) {
    int t = blockIdx.x * blockDim.x + threadIdx.x;
    int total = NN * NHEAD * HD;
    if (t >= total) return;
    int d = t % HD, h = (t / HD) % NHEAD, i = t / (HD * NHEAD);
    float o = 0.0f, l = 0.0f;
    for (int jc = 0; jc < NJC; ++jc) {
        o += Opart[(((size_t)jc * NN + i) * NHEAD + h) * HD + d];
        l += Lpart[((size_t)jc * NN + i) * NHEAD + h];
    }
    agg[(size_t)i * (NHEAD * HD) + h * HD + d] = o / fmaxf(l, 1e-30f);
}

// ---------------------------------------------------------------------------
// Output projection + residual (optional projection)
// ---------------------------------------------------------------------------
__global__ void oproj_kernel(const float* __restrict__ agg,
                             const float* __restrict__ Wo, const float* __restrict__ bo,
                             const float* __restrict__ xin, int INR,
                             const float* __restrict__ Wr, const float* __restrict__ br,
                             int HDIM, float* __restrict__ y) {
    int t = blockIdx.x * blockDim.x + threadIdx.x;
    if (t >= NN * HDIM) return;
    int i = t / HDIM, o = t % HDIM;
    float s = bo[o];
    for (int k = 0; k < HDIM; ++k) s += agg[(size_t)i * HDIM + k] * Wo[(size_t)k * HDIM + o];
    float r;
    if (Wr) {
        r = br[o];
        for (int k = 0; k < INR; ++k) r += xin[(size_t)i * INR + k] * Wr[(size_t)k * HDIM + o];
    } else {
        r = xin[(size_t)i * HDIM + o];
    }
    y[t] = s + r;
}

// ---------------------------------------------------------------------------
// LayerNorm (one wave per row) + optional ReLU
// ---------------------------------------------------------------------------
__global__ void ln_kernel(const float* __restrict__ y,
                          const float* __restrict__ ls, const float* __restrict__ lb,
                          int HDIM, int do_relu, float* __restrict__ xout) {
    int row  = blockIdx.x * (blockDim.x >> 5) + (threadIdx.x >> 5);
    int lane = threadIdx.x & 31;
    if (row >= NN) return;
    float s = 0.0f, s2 = 0.0f;
    for (int o = lane; o < HDIM; o += 32) {
        float v = y[(size_t)row * HDIM + o];
        s += v; s2 += v * v;
    }
#pragma unroll
    for (int m = 16; m >= 1; m >>= 1) {
        s  += __shfl_xor(s,  m, 32);
        s2 += __shfl_xor(s2, m, 32);
    }
    float mu  = s / HDIM;
    float var = s2 / HDIM - mu * mu;
    float rs  = rsqrtf(var + 1e-6f);
    for (int o = lane; o < HDIM; o += 32) {
        float v = (y[(size_t)row * HDIM + o] - mu) * rs * ls[o] + lb[o];
        if (do_relu) v = fmaxf(v, 0.0f);
        xout[(size_t)row * HDIM + o] = v;
    }
}

// ---------------------------------------------------------------------------
// Final heads: cbf = c3 + 0.1*(||nodes[:,:3]|| - 2);  g = g3 / max(||g3||, 1)
// d_out = [cbf (1024) | g (1024x6)]
// ---------------------------------------------------------------------------
__global__ void final_kernel(const float* __restrict__ nodes,
                             const float* __restrict__ c3o,
                             const float* __restrict__ g3o,
                             float* __restrict__ out) {
    int i = blockIdx.x * blockDim.x + threadIdx.x;
    if (i >= NN) return;
    float nx = nodes[i * 6 + 0], ny = nodes[i * 6 + 1], nz = nodes[i * 6 + 2];
    out[i] = c3o[i] + 0.1f * (sqrtf(nx * nx + ny * ny + nz * nz) - 2.0f);
    float g[6], gn = 0.0f;
#pragma unroll
    for (int k = 0; k < 6; ++k) { g[k] = g3o[i * 6 + k]; gn += g[k] * g[k]; }
    float inv = 1.0f / fmaxf(sqrtf(gn), 1.0f);
#pragma unroll
    for (int k = 0; k < 6; ++k) out[NN + i * 6 + k] = g[k] * inv;
}

// ---------------------------------------------------------------------------
// Host launcher.
// Input order (setup_inputs insertion order; params flattened with sorted dict
// keys, {'W','b'} -> W then b):
//  0 nodes  1 edges  2 adjacency
//  3..8   c1.W c1.b c2.W c2.b c3.W c3.b
//  9..14  g1.W g1.b g2.W g2.b g3.W g3.b
//  15..26 gat0: e.W e.b k.W k.b ln_b ln_s o.W o.b q.W q.b v.W v.b
//  27..38 gat1: (same)
//  39..52 gat2: e.W e.b k.W k.b ln_b ln_s o.W o.b q.W q.b r.W r.b v.W v.b
//  53..54 inp.W inp.b
// ---------------------------------------------------------------------------
extern "C" void kernel_launch(void* const* d_in, const int* in_sizes, int n_in,
                              void* d_out, int out_size, void* d_ws, size_t ws_size,
                              hipStream_t stream) {
    auto F = [&](int idx) { return (const float*)d_in[idx]; };
    const float* nodes = F(0);
    const float* edges = F(1);
    const int*   adj   = (const int*)d_in[2];

    // workspace bump allocator
    char* wsp = (char*)d_ws;
    auto alloc = [&](size_t bytes) -> void* {
        void* p = (void*)wsp;
        wsp += (bytes + 255) & ~(size_t)255;
        return p;
    };
    float*    xA    = (float*)alloc((size_t)NN * 128 * 4);
    float*    xB    = (float*)alloc((size_t)NN * 128 * 4);
    float*    qf    = (float*)alloc((size_t)NN * 128 * 4);
    _Float16* q16   = (_Float16*)alloc((size_t)NN * 128 * 2);
    _Float16* k16   = (_Float16*)alloc((size_t)NN * 128 * 2);
    _Float16* v16T  = (_Float16*)alloc((size_t)NN * 128 * 2);
    float*    tq    = (float*)alloc((size_t)NN * NHEAD * 8 * 4);
    float*    sbq   = (float*)alloc((size_t)NN * NHEAD * 4);
    float*    Opart = (float*)alloc((size_t)NJC * NN * 128 * 4);
    float*    Lpart = (float*)alloc((size_t)NJC * NN * NHEAD * 4);
    float*    agg   = (float*)alloc((size_t)NN * 128 * 4);
    float*    ybuf  = (float*)alloc((size_t)NN * 128 * 4);
    float*    h1    = (float*)alloc((size_t)NN * 64 * 4);
    float*    h2    = (float*)alloc((size_t)NN * 32 * 4);
    float*    c3o   = (float*)alloc((size_t)NN * 4);
    float*    g3o   = (float*)alloc((size_t)NN * 6 * 4);
    (void)ws_size; (void)n_in; (void)in_sizes; (void)out_size;

    auto grid = [](int n) { return (n + 255) / 256; };

    // Input embedding: x = relu(nodes @ Winp + binp)   [1024,6] -> [1024,128]
    dense_kernel<<<grid(NN * 128), 256, 0, stream>>>(nodes, F(53), F(54), 6, 128, 1, xA);

    const float* x = xA;
    float* xn = xB;
    const int gat_base[3] = {15, 27, 39};
    for (int l = 0; l < 3; ++l) {
        const int HDIM = (l < 2) ? 128 : 64;
        const int HD = HDIM / NHEAD;                  // 32, 32, 16
        const float scale = rsqrtf((float)HD);
        const int B = gat_base[l];
        const bool has_r = (l == 2);
        const float *eW = F(B+0), *eb = F(B+1), *kW = F(B+2), *kb = F(B+3);
        const float *lnb = F(B+4), *lns = F(B+5), *oW = F(B+6), *ob = F(B+7);
        const float *qW = F(B+8), *qb = F(B+9);
        const float *rW = has_r ? F(B+10) : nullptr;
        const float *rb = has_r ? F(B+11) : nullptr;
        const float *vW = has_r ? F(B+12) : F(B+10);
        const float *vb = has_r ? F(B+13) : F(B+11);

        qkv_kernel<<<grid(NN * HDIM), 256, 0, stream>>>(
            x, 128, qW, qb, kW, kb, vW, vb, HDIM, qf, q16, k16, v16T);
        tq_kernel<<<grid(NN * NHEAD), 256, 0, stream>>>(qf, eW, eb, HD, tq, sbq);

        // 64 i-tiles x NJC j-chunks waves, 8 waves per block
        const int attn_blocks = (64 * NJC) / 8;
        if (HD == 32) {
            gat_attn_kernel<32><<<attn_blocks, 256, 0, stream>>>(
                edges, adj, q16, k16, v16T, tq, sbq, scale, Opart, Lpart);
        } else {
            gat_attn_kernel<16><<<attn_blocks, 256, 0, stream>>>(
                edges, adj, q16, k16, v16T, tq, sbq, scale, Opart, Lpart);
        }
        gat_reduce_kernel<<<grid(NN * NHEAD * HD), 256, 0, stream>>>(Opart, Lpart, HD, agg);
        oproj_kernel<<<grid(NN * HDIM), 256, 0, stream>>>(
            agg, oW, ob, x, 128, rW, rb, HDIM, ybuf);
        ln_kernel<<<NN / 8, 256, 0, stream>>>(ybuf, lns, lnb, HDIM, (l < 2) ? 1 : 0, xn);

        const float* tmp = x; x = xn; xn = (float*)tmp;
    }

    // Heads (x is [1024,64])
    dense_kernel<<<grid(NN * 64), 256, 0, stream>>>(x,  F(3),  F(4),  64, 64, 1, h1);
    dense_kernel<<<grid(NN * 32), 256, 0, stream>>>(h1, F(5),  F(6),  64, 32, 1, h2);
    dense_kernel<<<grid(NN * 1),  256, 0, stream>>>(h2, F(7),  F(8),  32, 1,  0, c3o);
    dense_kernel<<<grid(NN * 64), 256, 0, stream>>>(x,  F(9),  F(10), 64, 64, 1, h1);
    dense_kernel<<<grid(NN * 32), 256, 0, stream>>>(h1, F(11), F(12), 64, 32, 1, h2);
    dense_kernel<<<grid(NN * 6),  256, 0, stream>>>(h2, F(13), F(14), 32, 6,  0, g3o);
    final_kernel<<<grid(NN), 256, 0, stream>>>(nodes, c3o, g3o, (float*)d_out);
}
